// DiffusionMACE_2370821947748
// MI455X (gfx1250) — matrix operand order repaired and checked by reading, not compiled, not run on hardware
//
#include <hip/hip_runtime.h>
#include <hip/hip_bf16.h>
#include <math.h>

#define N_NODES 10000
#define N_EDGES 160000
#define NBATCH  64
#define H_DIM   64
#define NE_ATTR 10
#define NIN_DIM 11
#define EPS_F   1e-9f

#define USE_ASYNC_STAGING 1

typedef __attribute__((ext_vector_type(16))) _Float16 v16h;
typedef __attribute__((ext_vector_type(8)))  _Float16 v8h;
typedef __attribute__((ext_vector_type(8)))  float    v8f;

// swizzled-weight pool offsets (halfs), per layer (stride 40960 halfs):
#define SW_WR2   0
#define SW_WUP   4096
#define SW_WMIX  8192
#define SW_WPROD 12288
#define SW_WRO1  16384
#define SW_WRO2  20480
#define SW_WG    24576
#define SW_WMLP1 28672
#define SW_WMLP2 36864
#define SW_LAYER 40960

__device__ __forceinline__ float silu_f(float x) {
  return x / (1.0f + __expf(-x));
}

// ---- WMMA fragment helpers (16x16x32 f16 -> f32), wave32 ----
// A: 16x32 tile from LDS row-major [16 rows x lda halfs], at column k0.
__device__ __forceinline__ v16h load_frag_a(const _Float16* A, int lda, int k0) {
  int lane = threadIdx.x & 31;
  int row  = lane & 15;
  int koff = (lane < 16) ? 0 : 16;
  const _Float16* p = A + row * lda + k0 + koff;
  v8h lo = *(const v8h*)(p);        // 16B aligned
  v8h hi = *(const v8h*)(p + 8);    // 16B aligned
  return __builtin_shufflevector(lo, hi, 0, 1, 2, 3, 4, 5, 6, 7,
                                 8, 9, 10, 11, 12, 13, 14, 15);
}

// B pre-swizzled: fragment (kt,nt), lane ls owns 16 contiguous halfs:
//   Bsw[(((kt*4+nt)*32+ls)<<4) + j] = B[kt*32 + ((ls&16)?16:0) + j][nt*16 + (ls&15)]
__device__ __forceinline__ v16h load_frag_b_sw(const _Float16* Bsw, int kt, int nt) {
  int lane = threadIdx.x & 31;
  return *(const v16h*)(Bsw + ((((kt * 4 + nt) * 32) + lane) << 4));
}

// stage pre-swizzled weights: global f16 -> LDS f16
#if USE_ASYNC_STAGING
// Async DMA path: GLOBAL_LOAD_ASYNC_TO_LDS_B128, tracked by ASYNCcnt.
// 16 bytes per lane per issue; LDS dest address = low 32 bits of generic pointer.
__device__ __forceinline__ void copy_w_lds(const _Float16* __restrict__ g,
                                           _Float16* lds, int nhalf, int tstride) {
  for (int s = threadIdx.x; s < (nhalf >> 3); s += tstride) {
    unsigned la = (unsigned)(size_t)(lds + (s << 3));
    const _Float16* gp = g + (s << 3);
    asm volatile("global_load_async_to_lds_b128 %0, %1, off"
                 :: "v"(la), "v"(gp) : "memory");
  }
}
__device__ __forceinline__ void wait_staging() {
  asm volatile("s_wait_asynccnt 0" ::: "memory");
}
#else
__device__ __forceinline__ void copy_w_lds(const _Float16* __restrict__ g,
                                           _Float16* lds, int nhalf, int tstride) {
  for (int s = threadIdx.x; s < (nhalf >> 4); s += tstride)
    *(v16h*)(lds + (s << 4)) = *(const v16h*)(g + (s << 4));
}
__device__ __forceinline__ void wait_staging() {}
#endif

// C/D: 16x16 f32; VGPR v holds row v (lanes 0-15) / row v+8 (lanes 16-31), col = lane&15.
__device__ __forceinline__ void store_frag_c(float* C, int ldc, int n0, v8f c) {
  int lane  = threadIdx.x & 31;
  int col   = n0 + (lane & 15);
  int rbase = (lane < 16) ? 0 : 8;
#pragma unroll
  for (int v = 0; v < 8; ++v) C[(rbase + v) * ldc + col] = c[v];
}

// C[RT*16 x 64] = A[RT*16 x K] @ B[K x 64]; (rt,nt) tiles distributed over waves.
__device__ __forceinline__ void mm_tiles(const _Float16* A, int lda, const _Float16* Bsw,
                                         float* C, int K, int RT, int wid, int nw) {
  int nkt = K >> 5;
  for (int t = wid; t < RT * 4; t += nw) {
    int rt = t >> 2, nt = t & 3;
    const _Float16* Ar = A + rt * 16 * lda;
    v8f acc = {};
    for (int kt = 0; kt < nkt; ++kt) {
      v16h a = load_frag_a(Ar, lda, kt * 32);
      v16h b = load_frag_b_sw(Bsw, kt, nt);
      acc = __builtin_amdgcn_wmma_f32_16x16x32_f16(false, a, false, b,
                                                   (short)0, acc, false, false);
    }
    store_frag_c(C + rt * 16 * 64, 64, nt * 16, acc);
  }
}

// ---- one-time weight pre-swizzle: f32 [K][64] -> swizzled f16 pool ----
__global__ void wconv_kernel(const float* __restrict__ Wr2, const float* __restrict__ Wup,
                             const float* __restrict__ Wmix, const float* __restrict__ Wprod,
                             const float* __restrict__ Wro1, const float* __restrict__ Wro2,
                             const float* __restrict__ Wg, const float* __restrict__ Wmlp1,
                             const float* __restrict__ Wmlp2, _Float16* __restrict__ out) {
  int s = blockIdx.x * blockDim.x + threadIdx.x;  // slot id, 16 halfs per slot
  if (s >= 2560) return;
  const float* W;
  int base;
  if      (s < 256)  { W = Wr2;   base = 0;    }
  else if (s < 512)  { W = Wup;   base = 256;  }
  else if (s < 768)  { W = Wmix;  base = 512;  }
  else if (s < 1024) { W = Wprod; base = 768;  }
  else if (s < 1280) { W = Wro1;  base = 1024; }
  else if (s < 1536) { W = Wro2;  base = 1280; }
  else if (s < 1792) { W = Wg;    base = 1536; }
  else if (s < 2304) { W = Wmlp1; base = 1792; }   // K = 128
  else               { W = Wmlp2; base = 2304; }
  int sl = s - base;
  int kt = sl >> 7, nt = (sl >> 5) & 3, ls = sl & 31;
  const float* src = W + (kt * 32 + ((ls & 16) ? 16 : 0)) * 64 + nt * 16 + (ls & 15);
  v16h v = {};
#pragma unroll
  for (int j = 0; j < 16; ++j) v[j] = (_Float16)src[j * 64];
  *(v16h*)(out + ((size_t)s << 4)) = v;
}

// ---- zero the batch-mean accumulators ----
__global__ void zero_kernel(float* sums, float* cnt) {
  int i = threadIdx.x;
  if (i < NBATCH * 3) sums[i] = 0.0f;
  if (i < NBATCH)     cnt[i]  = 0.0f;
}

// ---- h = [na/4, t[batch]] @ W_emb + b_emb ; f0 = h ; posw = positions ----
__global__ void embed_kernel(const float* __restrict__ na, const float* __restrict__ t,
                             const int* __restrict__ batch,
                             const float* __restrict__ W_emb, const float* __restrict__ b_emb,
                             const float* __restrict__ positions,
                             float* __restrict__ h, float* __restrict__ f0,
                             float* __restrict__ posw) {
  int tid = blockIdx.x * blockDim.x + threadIdx.x;
  if (tid < N_NODES * 3) posw[tid] = positions[tid];
  if (tid < N_NODES * H_DIM) {
    int n = tid >> 6, k = tid & 63;
    float acc = b_emb[k];
#pragma unroll
    for (int j = 0; j < NE_ATTR; ++j)
      acc += na[n * NE_ATTR + j] * 0.25f * W_emb[j * H_DIM + k];
    acc += t[batch[n]] * W_emb[NE_ATTR * H_DIM + k];
    h[tid]  = acc;
    f0[tid] = acc;
  }
}

// ---- per-layer prep: f0up = f0 @ W_up (WMMA), zero agg.
//      4 waves/block, each wave an independent 16-node tile; shared staged W_up.
__global__ void prep_kernel(const float* __restrict__ f0, const _Float16* __restrict__ WupSw,
                            float* __restrict__ f0up, float* __restrict__ agg) {
  __shared__ __align__(32) _Float16 sW[64 * 64];      // swizzled
  __shared__ __align__(32) _Float16 sX[4][16 * 64];
  __shared__ float sC[4][16 * 64];
  const int lane = threadIdx.x & 31;
  const int wid  = threadIdx.x >> 5;
  const size_t n0 = ((size_t)blockIdx.x * 4 + wid) * 16;
  const bool active = (n0 < N_NODES);

  copy_w_lds(WupSw, sW, 4096, 128);   // async issue; overlaps A staging
  if (active) {
    for (int i = lane; i < 16 * 64; i += 32) sX[wid][i] = (_Float16)f0[n0 * 64 + i];
  }
  wait_staging();
  __syncthreads();
  if (active) {
    mm_tiles(sX[wid], 64, sW, sC[wid], 64, 1, 0, 1);
    for (int i = lane; i < 16 * 64; i += 32) f0up[n0 * 64 + i] = sC[wid][i];
    for (int i = lane; i < 16 * 64 * 4; i += 32) agg[n0 * 256 + i] = 0.0f;
  }
}

// ---- edge kernel: 4 waves/block, 16 edges/wave (64/block); shared staged Wr2.
//      r = silu(len*Wr1+br1) @ Wr2 via WMMA; scatter agg[rcv,k,l] += r*f0up[snd,k]*Y[l]/16.
__global__ void edge_kernel(const int* __restrict__ ei, const float* __restrict__ shifts,
                            const float* __restrict__ pos, const float* __restrict__ f0up,
                            const float* __restrict__ Wr1, const float* __restrict__ br1,
                            const _Float16* __restrict__ Wr2Sw, float* __restrict__ agg) {
  __shared__ __align__(32) _Float16 sB[64 * 64];      // swizzled Wr2
  __shared__ __align__(32) _Float16 sA[4][16 * 64];
  __shared__ float sC[4][16 * 64];
  __shared__ float sY[4][16 * 4];
  __shared__ float sLen[4][16];
  __shared__ float sWr1[64], sbr1[64];
  __shared__ int   sSnd[4][16], sRcv[4][16];
  const int lane = threadIdx.x & 31;
  const int wid  = threadIdx.x >> 5;
  const int e0   = blockIdx.x * 64 + wid * 16;

  copy_w_lds(Wr2Sw, sB, 4096, 128);   // async issue; overlaps edge geometry
  if (threadIdx.x < 64) {
    sWr1[threadIdx.x] = Wr1[threadIdx.x];
    sbr1[threadIdx.x] = br1[threadIdx.x];
  }
  if (lane < 16) {
    int e = e0 + lane;
    int s = ei[e], r = ei[N_EDGES + e];
    sSnd[wid][lane] = s; sRcv[wid][lane] = r;
    float vx = pos[r * 3 + 0] - pos[s * 3 + 0] + shifts[e * 3 + 0];
    float vy = pos[r * 3 + 1] - pos[s * 3 + 1] + shifts[e * 3 + 1];
    float vz = pos[r * 3 + 2] - pos[s * 3 + 2] + shifts[e * 3 + 2];
    float len = sqrtf(vx * vx + vy * vy + vz * vz);
    sLen[wid][lane] = len;
    float inv = 1.0f / (len + EPS_F);
    sY[wid][lane * 4 + 0] = 1.0f;
    sY[wid][lane * 4 + 1] = 1.7320508f * vx * inv;
    sY[wid][lane * 4 + 2] = 1.7320508f * vy * inv;
    sY[wid][lane * 4 + 3] = 1.7320508f * vz * inv;
  }
  wait_staging();
  __syncthreads();
  for (int i = lane; i < 16 * 64; i += 32) {
    int e = i >> 6, hh = i & 63;
    sA[wid][i] = (_Float16)silu_f(sLen[wid][e] * sWr1[hh] + sbr1[hh]);
  }
  mm_tiles(sA[wid], 64, sB, sC[wid], 64, 1, 0, 1);
  const float inv_avg = 1.0f / 16.0f;  // 1/AVG_NEIGH
  for (int i = lane; i < 16 * 64; i += 32) {
    int e = i >> 6, k = i & 63;
    float mm = sC[wid][i] * f0up[(size_t)sSnd[wid][e] * 64 + k] * inv_avg;
    float* dst = agg + ((size_t)sRcv[wid][e] * 64 + k) * 4;
    atomicAdd(dst + 0, mm);
    atomicAdd(dst + 1, mm * sY[wid][e * 4 + 1]);
    atomicAdd(dst + 2, mm * sY[wid][e * 4 + 2]);
    atomicAdd(dst + 3, mm * sY[wid][e * 4 + 3]);
  }
}

// ---- node kernel: 128 threads cooperate on one 16-node group; all matmuls WMMA ----
__global__ void node_kernel(const float* __restrict__ agg,
                            const _Float16* __restrict__ wsw,  // layer's swizzled pool
                            const float* __restrict__ bro1, const float* __restrict__ wv,
                            const float* __restrict__ bmlp1, const float* __restrict__ bmlp2,
                            float* __restrict__ h, float* __restrict__ f0,
                            float* __restrict__ pos) {
  __shared__ __align__(32) _Float16 sW[128 * 64];  // swizzled weights (max K=128)
  __shared__ __align__(32) _Float16 sX[64 * 64];   // A operands
  __shared__ float sC[64 * 64];                    // a-rows, then nf-rows
  __shared__ float sT0[16 * 64];
  __shared__ float sScal[16 * 64];
  __shared__ float sGate[16 * 64];
  const int tid  = threadIdx.x;
  const int wid  = tid >> 5;
  const size_t n0 = (size_t)blockIdx.x * 16;

  // a = agg-rows @ W_mix  (rows r = n*4+l)
  copy_w_lds(wsw + SW_WMIX, sW, 4096, 128);
  for (int i = tid; i < 64 * 64; i += 128) {
    int r = i >> 6, hh = i & 63, n = r >> 2, l = r & 3;
    sX[i] = (_Float16)agg[((n0 + n) * 64 + hh) * 4 + l];
  }
  wait_staging();
  __syncthreads();
  mm_tiles(sX, 64, sW, sC, 64, 4, wid, 4);
  __syncthreads();
  // ap = a * (1 + s + s^2),  s = a[:,:,l=0]
  copy_w_lds(wsw + SW_WPROD, sW, 4096, 128);
  for (int i = tid; i < 64 * 64; i += 128) {
    int r = i >> 6, k = i & 63, n = r >> 2;
    float s = sC[(n * 4) * 64 + k];
    sX[i] = (_Float16)(sC[i] * (1.0f + s + s * s));
  }
  wait_staging();
  __syncthreads();
  mm_tiles(sX, 64, sW, sC, 64, 4, wid, 4);   // sC = nf rows
  __syncthreads();
  // sc = nf[:,:,0]
  copy_w_lds(wsw + SW_WRO1, sW, 4096, 128);
  for (int i = tid; i < 16 * 64; i += 128) {
    int n = i >> 6, k = i & 63;
    sX[i] = (_Float16)sC[(n * 4) * 64 + k];
  }
  wait_staging();
  __syncthreads();
  mm_tiles(sX, 64, sW, sT0, 64, 1, wid, 4);
  __syncthreads();
  copy_w_lds(wsw + SW_WRO2, sW, 4096, 128);
  for (int i = tid; i < 16 * 64; i += 128)
    sX[1024 + i] = (_Float16)silu_f(sT0[i] + bro1[i & 63]);
  wait_staging();
  __syncthreads();
  mm_tiles(sX + 1024, 64, sW, sScal, 64, 1, wid, 4);
  __syncthreads();
  copy_w_lds(wsw + SW_WG, sW, 4096, 128);
  wait_staging();
  __syncthreads();
  mm_tiles(sX, 64, sW, sT0, 64, 1, wid, 4);
  __syncthreads();
  for (int i = tid; i < 16 * 64; i += 128) sGate[i] = silu_f(sT0[i]);
  __syncthreads();
  // vout -> pos update (v3 = nf rows l=1..3)
  if (tid < 48) {
    int n = tid / 3, c = tid - n * 3;
    float acc = 0.0f;
    for (int hh = 0; hh < 64; ++hh)
      acc += sC[(n * 4 + 1 + c) * 64 + hh] * sGate[n * 64 + hh] * wv[hh];
    pos[(n0 + n) * 3 + c] += acc;
  }
  // f0 <- sc (feeds next layer's messages)
  for (int i = tid; i < 16 * 64; i += 128) {
    int n = i >> 6, k = i & 63;
    f0[(n0 + n) * 64 + k] = sC[(n * 4) * 64 + k];
  }
  __syncthreads();
  // h MLP: concat [h, scalars] (16x128) @ Wmlp1 -> silu -> @ Wmlp2
  copy_w_lds(wsw + SW_WMLP1, sW, 8192, 128);
  for (int i = tid; i < 16 * 64; i += 128) {
    int n = i >> 6, k = i & 63;
    sX[n * 128 + k]      = (_Float16)h[(n0 + n) * 64 + k];
    sX[n * 128 + 64 + k] = (_Float16)sScal[i];
  }
  wait_staging();
  __syncthreads();
  mm_tiles(sX, 128, sW, sT0, 128, 1, wid, 4);
  __syncthreads();
  copy_w_lds(wsw + SW_WMLP2, sW, 4096, 128);
  for (int i = tid; i < 16 * 64; i += 128)
    sX[2048 + i] = (_Float16)silu_f(sT0[i] + bmlp1[i & 63]);
  wait_staging();
  __syncthreads();
  mm_tiles(sX + 2048, 64, sW, sT0, 64, 1, wid, 4);
  __syncthreads();
  for (int i = tid; i < 16 * 64; i += 128) {
    int n = i >> 6, k = i & 63;
    h[(n0 + n) * 64 + k] += sT0[i] + bmlp2[k];
  }
}

// ---- per-batch position sums ----
__global__ void sums_kernel(const float* __restrict__ posw, const int* __restrict__ batch,
                            float* __restrict__ sums, float* __restrict__ cnt) {
  int n = blockIdx.x * blockDim.x + threadIdx.x;
  if (n < N_NODES) {
    int b = batch[n];
    atomicAdd(&sums[b * 3 + 0], posw[n * 3 + 0]);
    atomicAdd(&sums[b * 3 + 1], posw[n * 3 + 1]);
    atomicAdd(&sums[b * 3 + 2], posw[n * 3 + 2]);
    atomicAdd(&cnt[b], 1.0f);
  }
}

// ---- outputs: pred[:, :10] then pred_pos ----
__global__ void out_kernel(const float* __restrict__ h, const float* __restrict__ W_out,
                           const float* __restrict__ b_out, const float* __restrict__ posw,
                           const float* __restrict__ positions, const int* __restrict__ batch,
                           const float* __restrict__ sums, const float* __restrict__ cnt,
                           float* __restrict__ out) {
  int tid = blockIdx.x * blockDim.x + threadIdx.x;
  if (tid < N_NODES * NE_ATTR) {
    int n = tid / NE_ATTR, j = tid % NE_ATTR;
    float acc = b_out[j];
    for (int k = 0; k < H_DIM; ++k) acc += h[(size_t)n * 64 + k] * W_out[k * NIN_DIM + j];
    out[tid] = acc;
  }
  if (tid < N_NODES * 3) {
    int n = tid / 3;
    int b = batch[n];
    int c = tid - n * 3;
    float mean = sums[b * 3 + c] / fmaxf(cnt[b], 1.0f);
    out[N_NODES * NE_ATTR + tid] = posw[tid] - mean - positions[tid];
  }
}

extern "C" void kernel_launch(void* const* d_in, const int* in_sizes, int n_in,
                              void* d_out, int out_size, void* d_ws, size_t ws_size,
                              hipStream_t stream) {
  (void)in_sizes; (void)n_in; (void)out_size; (void)ws_size;
  const float* positions = (const float*)d_in[0];
  const float* node_attrs= (const float*)d_in[1];
  const float* t         = (const float*)d_in[2];
  const float* shifts    = (const float*)d_in[3];
  const int*   batch     = (const int*)d_in[4];
  const int*   edge_index= (const int*)d_in[5];
  const float* W_emb     = (const float*)d_in[6];
  const float* b_emb     = (const float*)d_in[7];
  const float* W_out     = (const float*)d_in[8];
  const float* b_out     = (const float*)d_in[9];
  const float* Wr1       = (const float*)d_in[10];
  const float* br1       = (const float*)d_in[11];
  const float* Wr2       = (const float*)d_in[12];
  const float* W_up      = (const float*)d_in[13];
  const float* W_mix     = (const float*)d_in[14];
  const float* W_prod    = (const float*)d_in[15];
  const float* Wro1      = (const float*)d_in[16];
  const float* bro1      = (const float*)d_in[17];
  const float* Wro2      = (const float*)d_in[18];
  const float* Wg        = (const float*)d_in[19];
  const float* wv        = (const float*)d_in[20];
  const float* Wmlp1     = (const float*)d_in[21];
  const float* bmlp1     = (const float*)d_in[22];
  const float* Wmlp2     = (const float*)d_in[23];
  const float* bmlp2     = (const float*)d_in[24];

  float* ws   = (float*)d_ws;
  float* h    = ws;                 // N*64
  float* f0   = ws + 640000;        // N*64
  float* f0up = ws + 1280000;       // N*64
  float* posw = ws + 1920000;       // N*3
  float* agg  = ws + 1950000;       // N*64*4
  float* sums = ws + 4510000;       // B*3
  float* cnt  = ws + 4510192;       // B
  _Float16* wsw = (_Float16*)(ws + 4520000);  // 2 layers * 40960 halfs, 32B-aligned
  float* out  = (float*)d_out;

  zero_kernel<<<1, 256, 0, stream>>>(sums, cnt);
  embed_kernel<<<(N_NODES * 64 + 255) / 256, 256, 0, stream>>>(
      node_attrs, t, batch, W_emb, b_emb, positions, h, f0, posw);
  for (int l = 0; l < 2; ++l) {
    wconv_kernel<<<20, 128, 0, stream>>>(
        Wr2 + l * 4096, W_up + l * 4096, W_mix + l * 4096, W_prod + l * 4096,
        Wro1 + l * 4096, Wro2 + l * 4096, Wg + l * 4096,
        Wmlp1 + l * 8192, Wmlp2 + l * 4096, wsw + l * SW_LAYER);
  }

  for (int l = 0; l < 2; ++l) {
    const _Float16* wl = wsw + l * SW_LAYER;
    prep_kernel<<<(N_NODES / 16 + 3) / 4, 128, 0, stream>>>(f0, wl + SW_WUP, f0up, agg);
    edge_kernel<<<N_EDGES / 64, 128, 0, stream>>>(
        edge_index, shifts, posw, f0up, Wr1 + l * 64, br1 + l * 64, wl + SW_WR2, agg);
    node_kernel<<<N_NODES / 16, 128, 0, stream>>>(
        agg, wl, bro1 + l * 64, wv + l * 64, bmlp1 + l * 64, bmlp2 + l * 64,
        h, f0, posw);
  }

  sums_kernel<<<(N_NODES + 255) / 256, 256, 0, stream>>>(posw, batch, sums, cnt);
  out_kernel<<<(N_NODES * NE_ATTR + 255) / 256, 256, 0, stream>>>(
      h, W_out, b_out, posw, positions, batch, sums, cnt, out);
}
